// TempConvLayer_45896020525815
// MI455X (gfx1250) — compile-verified, matrix operand order checked by reading, not run on hardware
//
#include <hip/hip_runtime.h>
#include <hip/hip_bf16.h>
#include <stdint.h>

// Problem constants (from reference)
#define C_DIM   64
#define T_DIM   20
#define B_DIM   2048
#define N_NODES (B_DIM * T_DIM)   // 40960
#define E_EDGES 1310720

typedef __attribute__((ext_vector_type(16))) __bf16 v16bf;
typedef __attribute__((ext_vector_type(8)))  float  v8f;

union BfPack {
  uint32_t u[8];
  uint4    q[2];
  v16bf    v;
};

// Native packed f32 -> bf16 conversion (RNE), one VOP3 op for two values.
__device__ __forceinline__ uint32_t cvt_pk_bf16(float lo, float hi) {
  uint32_t d;
  asm("v_cvt_pk_bf16_f32 %0, %1, %2" : "=v"(d) : "v"(lo), "v"(hi));
  return d;
}

// CDNA5 native float max atomic, no return (STOREcnt-tracked).
// nt selects a compile-time byte offset folded into the instruction.
__device__ __forceinline__ void atomic_fmax_o(float* a, float v, int nt) {
  switch (nt) {
  case 0: asm volatile("global_atomic_max_num_f32 %0, %1, off"
                       :: "v"(a), "v"(v) : "memory"); break;
  case 1: asm volatile("global_atomic_max_num_f32 %0, %1, off offset:64"
                       :: "v"(a), "v"(v) : "memory"); break;
  case 2: asm volatile("global_atomic_max_num_f32 %0, %1, off offset:128"
                       :: "v"(a), "v"(v) : "memory"); break;
  default: asm volatile("global_atomic_max_num_f32 %0, %1, off offset:192"
                       :: "v"(a), "v"(v) : "memory"); break;
  }
}

// Packed bf16 leaky-relu(a+b): leaky(x) = max(x, slope*x)  (slope in (0,1))
__device__ __forceinline__ uint32_t pk_leaky_add(uint32_t a, uint32_t b, uint32_t slope2) {
  uint32_t s, u, r;
  asm("v_pk_add_bf16 %0, %1, %2" : "=v"(s) : "v"(a), "v"(b));
  asm("v_pk_mul_bf16 %0, %1, %2" : "=v"(u) : "v"(s), "v"(slope2));
  asm("v_pk_max_num_bf16 %0, %1, %2" : "=v"(r) : "v"(s), "v"(u));
  return r;
}

// ---------------------------------------------------------------------------
// Pack BF16 weight B-tiles in the ISA 16-bit B-matrix (32x16) lane layout.
// Tile t = ntile*2 + kslice. Lane l holds col n = ntile*16 + (l&15),
// K = kslice*32 + 8*(l>>4) + {0..7, 16..23}, pairs packed lo|hi per dword.
// w==0: Wdiff = W1[:, :64] - W1[:, 64:]   (for P)
// w==1: W1b   = W1[:, 64:]               (for Q)
// w==2: W2
// ---------------------------------------------------------------------------
__global__ void prep_weights_kernel(const float* __restrict__ W1,
                                    const float* __restrict__ W2,
                                    uint32_t* __restrict__ wp) {
  int g = blockIdx.x * blockDim.x + threadIdx.x;
  if (g >= 3 * 8 * 32) return;
  int w    = g / 256;
  int r    = g % 256;
  int t    = r / 32;
  int lane = r % 32;
  int ntile = t >> 1, ks = t & 1;
  int n     = ntile * 16 + (lane & 15);
  int kbase = ks * 32 + ((lane >> 4) << 3);
  float vals[16];
#pragma unroll
  for (int j = 0; j < 16; ++j) {
    int k = kbase + (j < 8 ? j : (j + 8));   // {0..7} then {16..23}
    float f;
    if (w == 0)      f = W1[n * 128 + k] - W1[n * 128 + 64 + k];
    else if (w == 1) f = W1[n * 128 + 64 + k];
    else             f = W2[n * 64 + k];
    vals[j] = f;
  }
  uint32_t* d = wp + (((w * 8 + t) * 32) + lane) * 8;
#pragma unroll
  for (int j = 0; j < 8; ++j) d[j] = cvt_pk_bf16(vals[2 * j], vals[2 * j + 1]);
}

__global__ void init_agg_kernel(float* __restrict__ agg, int n) {
  int i = blockIdx.x * blockDim.x + threadIdx.x;
  if (i < n) agg[i] = __uint_as_float(0xFF800000u);  // -inf
}

__device__ __forceinline__ void load_wtile(const uint32_t* __restrict__ wp,
                                           int w, int t, int lane, BfPack& out) {
  const uint4* s = (const uint4*)(wp + (((w * 8 + t) * 32) + lane) * 8);
  out.q[0] = s[0];
  out.q[1] = s[1];
}

// ---------------------------------------------------------------------------
// Node transform: Pb = bf16(x @ Wdiff^T + b1) ; Qb = bf16(x @ W1b^T)
// One wave handles 16 nodes. Outputs stored bf16 [node][channel] so the edge
// kernel gathers half the bytes and needs no conversion before its WMMAs.
// ---------------------------------------------------------------------------
__global__ void node_transform_kernel(const float* __restrict__ x,
                                      const uint32_t* __restrict__ wp,
                                      const float* __restrict__ b1,
                                      uint16_t* __restrict__ Pb,
                                      uint16_t* __restrict__ Qb) {
  int wave = (blockIdx.x * blockDim.x + threadIdx.x) >> 5;
  int lane = threadIdx.x & 31;
  if (wave >= N_NODES / 16) return;
  int m = lane & 15, half = lane >> 4;

  const float* xr = x + (size_t)(wave * 16 + m) * C_DIM;
  BfPack a0p, a1p;
#pragma unroll
  for (int c = 0; c < 4; ++c) {
    const float4* p = (const float4*)(xr + half * 8 + c * 16);
    float4 lo = p[0], hi = p[1];
    BfPack& dp = (c < 2) ? a0p : a1p;
    int o = (c & 1) * 4;
    dp.u[o + 0] = cvt_pk_bf16(lo.x, lo.y);
    dp.u[o + 1] = cvt_pk_bf16(lo.z, lo.w);
    dp.u[o + 2] = cvt_pk_bf16(hi.x, hi.y);
    dp.u[o + 3] = cvt_pk_bf16(hi.z, hi.w);
  }

#pragma unroll
  for (int nt = 0; nt < 4; ++nt) {
    int n = nt * 16 + m;
    float bias = b1[n];
    {
      BfPack w0, w1;
      load_wtile(wp, 0, nt * 2 + 0, lane, w0);
      load_wtile(wp, 0, nt * 2 + 1, lane, w1);
      v8f acc = {};
      acc = __builtin_amdgcn_wmma_f32_16x16x32_bf16(false, a0p.v, false, w0.v,
                                                    (short)0, acc, false, false);
      acc = __builtin_amdgcn_wmma_f32_16x16x32_bf16(false, a1p.v, false, w1.v,
                                                    (short)0, acc, false, false);
#pragma unroll
      for (int r = 0; r < 8; ++r) {
        float v = acc[r] + bias;
        Pb[(size_t)(wave * 16 + r + half * 8) * C_DIM + n] =
            (uint16_t)cvt_pk_bf16(v, v);       // store_b16 reads VDATA[15:0]
      }
    }
    {
      BfPack w0, w1;
      load_wtile(wp, 1, nt * 2 + 0, lane, w0);
      load_wtile(wp, 1, nt * 2 + 1, lane, w1);
      v8f acc = {};
      acc = __builtin_amdgcn_wmma_f32_16x16x32_bf16(false, a0p.v, false, w0.v,
                                                    (short)0, acc, false, false);
      acc = __builtin_amdgcn_wmma_f32_16x16x32_bf16(false, a1p.v, false, w1.v,
                                                    (short)0, acc, false, false);
#pragma unroll
      for (int r = 0; r < 8; ++r) {
        float v = acc[r];
        Qb[(size_t)(wave * 16 + r + half * 8) * C_DIM + n] =
            (uint16_t)cvt_pk_bf16(v, v);
      }
    }
  }
}

// ---------------------------------------------------------------------------
// Edge MLP: h = leaky(Pb[dst] + Qb[src]) in packed bf16; msg = h @ W2^T + b2
// via WMMA; scatter with native float-max atomics (offsets folded into the
// instruction immediate). One wave = 16 edges = one 16x64 A-tile; 8 WMMAs.
// ---------------------------------------------------------------------------
__global__ void edge_mlp_kernel(const int* __restrict__ eidx,
                                const uint16_t* __restrict__ Pb,
                                const uint16_t* __restrict__ Qb,
                                const uint32_t* __restrict__ wp,
                                const float* __restrict__ b2,
                                float* __restrict__ agg) {
  int wave = (blockIdx.x * blockDim.x + threadIdx.x) >> 5;
  int lane = threadIdx.x & 31;
  if (wave >= E_EDGES / 16) return;
  int m = lane & 15, half = lane >> 4;

  int dst = eidx[wave * 16 + m];
  int src = eidx[E_EDGES + wave * 16 + m];

  // dst indices for D-matrix rows half*8 + 0..7: 8 consecutive ints (cached)
  const int4* drp = (const int4*)(eidx + wave * 16 + half * 8);
  int4 dr0 = drp[0], dr1 = drp[1];
  int drow[8] = {dr0.x, dr0.y, dr0.z, dr0.w, dr1.x, dr1.y, dr1.z, dr1.w};

  const uint32_t slope2 = 0x3C243C24u;  // bf16(0.01) x2

  const uint32_t* pr = (const uint32_t*)(Pb + (size_t)dst * C_DIM); // 32 dw/row
  const uint32_t* qr = (const uint32_t*)(Qb + (size_t)src * C_DIM);

  BfPack a0p, a1p;
#pragma unroll
  for (int c = 0; c < 4; ++c) {
    int dbase = half * 4 + c * 8;                 // dword index of 8-ch chunk
    uint4 pq = *(const uint4*)(pr + dbase);
    uint4 qq = *(const uint4*)(qr + dbase);
    BfPack& dp = (c < 2) ? a0p : a1p;
    int o = (c & 1) * 4;
    dp.u[o + 0] = pk_leaky_add(pq.x, qq.x, slope2);
    dp.u[o + 1] = pk_leaky_add(pq.y, qq.y, slope2);
    dp.u[o + 2] = pk_leaky_add(pq.z, qq.z, slope2);
    dp.u[o + 3] = pk_leaky_add(pq.w, qq.w, slope2);
  }

  BfPack w[8];
#pragma unroll
  for (int t = 0; t < 8; ++t) load_wtile(wp, 2, t, lane, w[t]);

  v8f acc0 = {}, acc1 = {}, acc2 = {}, acc3 = {};
  acc0 = __builtin_amdgcn_wmma_f32_16x16x32_bf16(false, a0p.v, false, w[0].v, (short)0, acc0, false, false);
  acc0 = __builtin_amdgcn_wmma_f32_16x16x32_bf16(false, a1p.v, false, w[1].v, (short)0, acc0, false, false);
  acc1 = __builtin_amdgcn_wmma_f32_16x16x32_bf16(false, a0p.v, false, w[2].v, (short)0, acc1, false, false);
  acc1 = __builtin_amdgcn_wmma_f32_16x16x32_bf16(false, a1p.v, false, w[3].v, (short)0, acc1, false, false);
  acc2 = __builtin_amdgcn_wmma_f32_16x16x32_bf16(false, a0p.v, false, w[4].v, (short)0, acc2, false, false);
  acc2 = __builtin_amdgcn_wmma_f32_16x16x32_bf16(false, a1p.v, false, w[5].v, (short)0, acc2, false, false);
  acc3 = __builtin_amdgcn_wmma_f32_16x16x32_bf16(false, a0p.v, false, w[6].v, (short)0, acc3, false, false);
  acc3 = __builtin_amdgcn_wmma_f32_16x16x32_bf16(false, a1p.v, false, w[7].v, (short)0, acc3, false, false);

  float b2v0 = b2[m], b2v1 = b2[16 + m], b2v2 = b2[32 + m], b2v3 = b2[48 + m];

#pragma unroll
  for (int r = 0; r < 8; ++r) {
    float* base = agg + (size_t)drow[r] * C_DIM + m;  // col m; +16 per nt via imm
    atomic_fmax_o(base, acc0[r] + b2v0, 0);
    atomic_fmax_o(base, acc1[r] + b2v1, 1);
    atomic_fmax_o(base, acc2[r] + b2v2, 2);
    atomic_fmax_o(base, acc3[r] + b2v3, 3);
  }
}

__global__ void finalize_kernel(const float* __restrict__ agg,
                                const float* __restrict__ x,
                                float* __restrict__ out, int n) {
  int i = blockIdx.x * blockDim.x + threadIdx.x;
  if (i < n) {
    float a = agg[i];
    if (__float_as_uint(a) == 0xFF800000u) a = 0.0f;  // empty segment -> 0
    float o = a + x[i];
    out[i] = o > 0.0f ? o : 0.0f;
  }
}

extern "C" void kernel_launch(void* const* d_in, const int* in_sizes, int n_in,
                              void* d_out, int out_size, void* d_ws, size_t ws_size,
                              hipStream_t stream) {
  const float* x    = (const float*)d_in[0];
  const int*   eidx = (const int*)d_in[1];   // [2, E]
  const float* W1   = (const float*)d_in[2]; // [64, 128]
  const float* b1   = (const float*)d_in[3];
  const float* W2   = (const float*)d_in[4]; // [64, 64]
  const float* b2   = (const float*)d_in[5];
  float* out = (float*)d_out;

  // Workspace layout (256B aligned): Pb, Qb (bf16 [N,64]), agg (f32 [N,64]),
  // packed bf16 weight tiles.
  uint16_t* Pb   = (uint16_t*)d_ws;
  uint16_t* Qb   = Pb + (size_t)N_NODES * C_DIM;
  float*    agg  = (float*)(Qb + (size_t)N_NODES * C_DIM);
  uint32_t* wpck = (uint32_t*)(agg + (size_t)N_NODES * C_DIM);

  const int nelem = N_NODES * C_DIM;          // 2,621,440

  prep_weights_kernel<<<3, 256, 0, stream>>>(W1, W2, wpck);
  init_agg_kernel<<<(nelem + 255) / 256, 256, 0, stream>>>(agg, nelem);
  node_transform_kernel<<<(N_NODES / 16) * 32 / 256, 256, 0, stream>>>(x, wpck, b1, Pb, Qb);
  edge_mlp_kernel<<<(E_EDGES / 16) * 32 / 256, 256, 0, stream>>>(eidx, Pb, Qb, wpck, b2, agg);
  finalize_kernel<<<(nelem + 255) / 256, 256, 0, stream>>>(agg, x, out, nelem);
}
